// DSTM_46557445488854
// MI455X (gfx1250) — matrix operand order repaired and checked by reading, compiled, not run on hardware
//
#include <hip/hip_runtime.h>
#include <hip/hip_bf16.h>

// ---------------- problem constants ----------------
#define BB 32
#define TT 2048
#define DD 128
#define MM 256
#define NROWS (BB * TT)          // 65536
#define ALPHA_INV_F (1.0f / (128.0f * 100.0f))

typedef __attribute__((ext_vector_type(2)))  float    v2f;
typedef __attribute__((ext_vector_type(8)))  float    v8f;
typedef __attribute__((ext_vector_type(16))) __bf16   v16bf;
typedef __attribute__((ext_vector_type(4)))  unsigned v4u;
typedef __attribute__((ext_vector_type(8)))  int      v8i;
typedef __attribute__((ext_vector_type(4)))  int      v4i;

// low 32 bits of a generic pointer to LDS == byte offset in group segment
__device__ __forceinline__ unsigned lds_addr32(const void* p) {
    return (unsigned)(unsigned long long)p;
}

// ---------------------------------------------------------------------
// Tensor Data Mover: dense 2D tile (rows x width elems, row-major with
// row_stride) global -> LDS.  Descriptor per CDNA5 ISA ch.8 (D# groups).
// dsz_log2: 0=1B 1=2B 2=4B.  (6-arg builtin: g0,g1,g2,g3,g4,cpol)
// ---------------------------------------------------------------------
__device__ __forceinline__ void tdm_load_2d(unsigned lds_off, const void* gptr,
                                            unsigned width, unsigned rows,
                                            unsigned row_stride, unsigned dsz_log2)
{
    unsigned long long ga = (unsigned long long)gptr;
    v4u g0; v8i g1;
    v4i gz4 = {0, 0, 0, 0};
    v8i gz8 = {0, 0, 0, 0, 0, 0, 0, 0};
    g0[0] = 1u;                                                 // count=1, user mode
    g0[1] = lds_off;                                            // lds_addr
    g0[2] = (unsigned)ga;                                       // global_addr lo
    g0[3] = (unsigned)((ga >> 32) & 0x01FFFFFFu) | (2u << 30);  // addr hi | type=2
    g1[0] = (int)(dsz_log2 << 16);                              // wg_mask=0, data_size
    g1[1] = (int)((width & 0xFFFFu) << 16);                     // tensor_dim0[15:0]
    g1[2] = (int)(((width >> 16) & 0xFFFFu) |                   // tensor_dim0[31:16]
                  ((rows & 0xFFFFu) << 16));                    // tensor_dim1[15:0]
    g1[3] = (int)(((rows >> 16) & 0xFFFFu) |                    // tensor_dim1[31:16]
                  ((width & 0xFFFFu) << 16));                   // tile_dim0
    g1[4] = (int)(rows & 0xFFFFu);                              // tile_dim1, tile_dim2=0
    g1[5] = (int)row_stride;                                    // tensor_dim0_stride lo
    g1[6] = 0;
    g1[7] = 0;
    __builtin_amdgcn_tensor_load_to_lds(g0, g1, gz4, gz4, gz8, 0);
}

// =====================================================================
// Setup: batch f32 -> bf16 ; weights [K,Cin,256] f32 -> bf16 [K,256,Cin]
// =====================================================================
__global__ void __launch_bounds__(256)
cvt_f32_bf16(const float* __restrict__ in, __bf16* __restrict__ out, int n)
{
    const int i = blockIdx.x * 256 + threadIdx.x;
    if (i < n) out[i] = (__bf16)in[i];
}

__global__ void __launch_bounds__(256)
transpose_w(const float* __restrict__ w, __bf16* __restrict__ wT, int cin, int n)
{
    const int i = blockIdx.x * 256 + threadIdx.x;   // i = (tap*256+co)*cin + ci
    if (i >= n) return;
    const int ci = i % cin;
    const int rest = i / cin;
    const int co = rest & 255;
    const int tap = rest >> 8;
    wT[i] = (__bf16)w[((size_t)tap * cin + ci) * 256 + co];
}

// =====================================================================
// Conv layer, bf16 WMMA.  One workgroup = one 16-timestep block x 256
// couts (8 waves x 2 cout tiles).  X tile staged via TDM into LDS with
// 4-row causal halo; weights pre-transposed bf16 [tap][cout][cin].
// =====================================================================
template<typename TOUT, int NTAPS, int CIN, bool RELU>
__global__ void __launch_bounds__(256)
conv_wmma(const __bf16* __restrict__ x, const __bf16* __restrict__ wT,
          const float* __restrict__ bias, TOUT* __restrict__ y, int dil)
{
    constexpr int COUT = 256;
    constexpr int HALO = 4;                       // >= max dilation
    __shared__ __bf16 Xt[(16 + HALO) * CIN];
    __shared__ TOUT   Yt[16 * COUT];

    const int tid  = threadIdx.x;
    const int lane = tid & 31, wv = tid >> 5;
    const int ln   = lane & 15, hi = lane >> 4;
    const int blk  = blockIdx.x;                  // 4096 blocks
    const int b    = blk / (TT / 16);
    const int t0   = (blk % (TT / 16)) * 16;

    // zero causal halo at batch start; TDM the rest of the X tile
    if (t0 == 0) {
        for (int i = tid; i < HALO * CIN; i += 256) Xt[i] = (__bf16)0.0f;
    }
    if (wv == 0) {
        const int tstart = (t0 == 0) ? 0 : (t0 - HALO);
        const int nrows  = (t0 == 0) ? 16 : (16 + HALO);
        const unsigned ldsoff =
            lds_addr32(&Xt[0]) + ((t0 == 0) ? (unsigned)(HALO * CIN * 2) : 0u);
        tdm_load_2d(ldsoff, x + ((size_t)b * TT + tstart) * CIN, CIN, nrows, CIN, 1);
        __builtin_amdgcn_s_wait_tensorcnt(0);
    }
    __syncthreads();

    const int c0 = wv * 32;                       // this wave's cout base (2 tiles)
    v8f acc0 = {}, acc1 = {};
    #pragma unroll
    for (int tap = 0; tap < NTAPS; ++tap) {
        const int off = (NTAPS - 1 - tap) * dil;  // causal tap offset
        const __bf16* xrow = &Xt[(ln + HALO - off) * CIN];
        const __bf16* w0r = wT + ((size_t)tap * COUT + c0 + ln) * CIN;
        const __bf16* w1r = w0r + (size_t)16 * CIN;
        #pragma unroll
        for (int kc = 0; kc < CIN / 32; ++kc) {
            v16bf a, b0, b1;
            #pragma unroll
            for (int i = 0; i < 16; ++i) {        // A 16x32: contiguous k = 16*hi + i (B uses same order)
                const int k = hi * 16 + i;
                a[i] = xrow[kc * 32 + k];
            }
            #pragma unroll
            for (int i = 0; i < 16; ++i) {        // B 32x16: same contiguous k = 16*hi + i as A
                const int k = hi * 16 + i;
                b0[i] = w0r[kc * 32 + k];
                b1[i] = w1r[kc * 32 + k];
            }
            acc0 = __builtin_amdgcn_wmma_f32_16x16x32_bf16(
                       false, a, false, b0, (short)0, acc0, false, false);
            acc1 = __builtin_amdgcn_wmma_f32_16x16x32_bf16(
                       false, a, false, b1, (short)0, acc1, false, false);
        }
    }
    const float bv0 = bias[c0 + ln], bv1 = bias[c0 + 16 + ln];
    #pragma unroll
    for (int r = 0; r < 8; ++r) {
        const int t = r + hi * 8;                 // C layout: M = r + 8*hi
        float v0 = acc0[r] + bv0, v1 = acc1[r] + bv1;
        if (RELU) { v0 = fmaxf(v0, 0.0f); v1 = fmaxf(v1, 0.0f); }
        Yt[t * COUT + c0 + ln]      = (TOUT)v0;
        Yt[t * COUT + c0 + 16 + ln] = (TOUT)v1;
    }
    __syncthreads();
    for (int i = tid; i < 16 * COUT; i += 256)    // coalesced 256-wide rows
        y[((size_t)b * TT + t0 + (i >> 8)) * COUT + (i & 255)] = Yt[i];
}

// =====================================================================
// Softmax over M=256, in place. One block per row.
// =====================================================================
__global__ void __launch_bounds__(256)
softmax256(float* __restrict__ p)
{
    __shared__ float red[256];
    const int m = threadIdx.x;
    float* row = p + (size_t)blockIdx.x * MM;
    const float v = row[m];
    red[m] = v; __syncthreads();
    for (int s = 128; s > 0; s >>= 1) {
        if (m < s) red[m] = fmaxf(red[m], red[m + s]);
        __syncthreads();
    }
    const float mx = red[0]; __syncthreads();
    const float e = __expf(v - mx);               // TAU = 1
    red[m] = e; __syncthreads();
    for (int s = 128; s > 0; s >>= 1) {
        if (m < s) red[m] += red[m + s];
        __syncthreads();
    }
    row[m] = e / red[0];
}

// =====================================================================
// Fast-weight scan, chunked (C=16) linear-attention form, f32 WMMA,
// W resident in LDS (128KB), TDM double-buffered H/S prefetch.
//   num = H*W0 + strict_tril(H*H^T)*S ;  prob = num / rowsum(num)
//   W  += H^T * S
// One workgroup (8 waves) per batch; wave w owns d-cols [16w,16w+16).
// =====================================================================
__global__ void __launch_bounds__(256)
scan_fw(const float* __restrict__ hsrc, const float* __restrict__ s_in,
        float* __restrict__ probs, float* __restrict__ Wout)
{
    __shared__ float Wl[MM * DD];     // 131072 B
    __shared__ float Hb[2][16 * MM];  //  32768 B (double buffer)
    __shared__ float Sb[2][16 * DD];  //  16384 B (double buffer)
    __shared__ float Nl[16 * DD];     //   8192 B
    __shared__ float Al[16 * 16];     //   1024 B
    __shared__ float rs[16];

    const int b    = blockIdx.x;
    const int tid  = threadIdx.x;
    const int lane = tid & 31, wv = tid >> 5;
    const int ln   = lane & 15, hi = lane >> 4;

    for (int i = tid; i < MM * DD; i += 256) Wl[i] = ALPHA_INV_F;

    if (wv == 0) {                                 // prefetch chunk 0
        tdm_load_2d(lds_addr32(&Hb[0][0]), hsrc + (size_t)b * TT * MM, MM, 16, MM, 2);
        tdm_load_2d(lds_addr32(&Sb[0][0]), s_in + (size_t)b * TT * DD, DD, 16, DD, 2);
    }
    __syncthreads();

    for (int cc = 0; cc < TT / 16; ++cc) {
        const int t0  = cc * 16;
        const int buf = cc & 1;
        const float* Hl = Hb[buf];
        const float* Sl = Sb[buf];

        // wave 0: issue prefetch of next chunk, retire current chunk's pair
        if (wv == 0) {
            if (cc + 1 < TT / 16) {
                const int t1 = t0 + 16;
                tdm_load_2d(lds_addr32(&Hb[buf ^ 1][0]),
                            hsrc + ((size_t)b * TT + t1) * MM, MM, 16, MM, 2);
                tdm_load_2d(lds_addr32(&Sb[buf ^ 1][0]),
                            s_in + ((size_t)b * TT + t1) * DD, DD, 16, DD, 2);
                __builtin_amdgcn_s_wait_tensorcnt(2);   // in-order: current pair done
            } else {
                __builtin_amdgcn_s_wait_tensorcnt(0);
            }
        }
        __syncthreads();

        // ---- wave 0: A = H * H^T (16x16, K=256 via 64 K4 WMMAs) ----
        if (wv == 0) {
            v8f a = {};
            for (int kc = 0; kc < MM / 4; ++kc) {
                v2f fa, fb;
                #pragma unroll
                for (int i = 0; i < 2; ++i) {
                    const int m = kc * 4 + hi * 2 + i;   // f32 A/B K layout
                    fa[i] = Hl[ln * MM + m];             // A row t   = ln
                    fb[i] = Hl[ln * MM + m];             // B col tau = ln
                }
                a = __builtin_amdgcn_wmma_f32_16x16x4_f32(
                        false, fa, false, fb, (short)0, a, false, false);
            }
            #pragma unroll
            for (int r = 0; r < 8; ++r) Al[(r + hi * 8) * 16 + ln] = a[r];
        }

        // ---- all waves: Num = H * W0 (this wave's 16 d-columns) ----
        v8f nacc = {};
        for (int kc = 0; kc < MM / 4; ++kc) {
            v2f fa, fb;
            #pragma unroll
            for (int i = 0; i < 2; ++i) {
                const int m = kc * 4 + hi * 2 + i;
                fa[i] = Hl[ln * MM + m];
                fb[i] = Wl[m * DD + wv * 16 + ln];
            }
            nacc = __builtin_amdgcn_wmma_f32_16x16x4_f32(
                       false, fa, false, fb, (short)0, nacc, false, false);
        }
        __syncthreads();   // A ready; all W reads complete

        // ---- strict-causal intra-chunk term, write Num ----
        {
            const int d = wv * 16 + ln;
            #pragma unroll
            for (int r = 0; r < 8; ++r) {
                const int t = r + hi * 8;
                float v = nacc[r];
                for (int tau = 0; tau < 16; ++tau)
                    if (tau < t) v += Al[t * 16 + tau] * Sl[tau * DD + d];
                Nl[t * DD + d] = v;
            }
        }
        __syncthreads();

        // ---- rowsum + normalize-after + emit probs ----
        if (tid < 16) {
            float s = 0.0f;
            for (int d = 0; d < DD; ++d) s += Nl[tid * DD + d];
            rs[tid] = s;
        }
        __syncthreads();
        for (int i = tid; i < 16 * DD; i += 256) {
            const int t = i >> 7, d = i & 127;
            probs[((size_t)b * TT + t0 + t) * DD + d] = Nl[i] / rs[t];
        }

        // ---- W += H^T * S (rank-16 update, 16 m-tiles x 4 K4 WMMAs) ----
        {
            const int d = wv * 16 + ln;
            for (int mb = 0; mb < 16; ++mb) {
                v8f wacc;
                #pragma unroll
                for (int r = 0; r < 8; ++r)
                    wacc[r] = Wl[(mb * 16 + r + hi * 8) * DD + d];
                #pragma unroll
                for (int kc = 0; kc < 4; ++kc) {
                    v2f fa, fb;
                    #pragma unroll
                    for (int i = 0; i < 2; ++i) {
                        const int t = kc * 4 + hi * 2 + i;
                        fa[i] = Hl[t * MM + mb * 16 + ln];  // H^T(m=ln, t)
                        fb[i] = Sl[t * DD + d];             // S(t, d)
                    }
                    wacc = __builtin_amdgcn_wmma_f32_16x16x4_f32(
                               false, fa, false, fb, (short)0, wacc, false, false);
                }
                #pragma unroll
                for (int r = 0; r < 8; ++r)
                    Wl[(mb * 16 + r + hi * 8) * DD + d] = wacc[r];
            }
        }
        __syncthreads();   // all reads of this chunk done before buffer reuse
    }

    for (int i = tid; i < MM * DD; i += 256)
        Wout[(size_t)b * MM * DD + i] = Wl[i];
}

// =====================================================================
// Launcher.  Workspace layout (bytes), total ~113 MB:
//   0        wT0  (2*128*256 bf16)      131072
//   131072   wT1  (2*256*256 bf16)      262144
//   393216   wT2  (2*256*256 bf16)      262144
//   655360   wTo  (1*256*256 bf16)      131072
//   786432   xb   (NROWS*128 bf16)    16777216
//   17563648 h0   (NROWS*256 bf16)    33554432   (h2 reuses this region)
//   51118080 h1   (NROWS*256 bf16)    33554432
//   51118080 logits (NROWS*256 f32)   67108864   (overwrites dead h1)
// =====================================================================
extern "C" void kernel_launch(void* const* d_in, const int* in_sizes, int n_in,
                              void* d_out, int out_size, void* d_ws, size_t ws_size,
                              hipStream_t stream)
{
    const float* batch = (const float*)d_in[0];
    const float* w0 = (const float*)d_in[1];  const float* b0 = (const float*)d_in[2];
    const float* w1 = (const float*)d_in[3];  const float* b1 = (const float*)d_in[4];
    const float* w2 = (const float*)d_in[5];  const float* b2 = (const float*)d_in[6];
    const float* wo = (const float*)d_in[7];  const float* bo = (const float*)d_in[8];

    float* probs = (float*)d_out;                        // [B,T,128]
    float* Wfin  = probs + (size_t)BB * TT * DD;         // [B,256,128]

    char* ws = (char*)d_ws;
    __bf16* wT0 = (__bf16*)(ws + 0);
    __bf16* wT1 = (__bf16*)(ws + 131072);
    __bf16* wT2 = (__bf16*)(ws + 393216);
    __bf16* wTo = (__bf16*)(ws + 655360);
    __bf16* xb  = (__bf16*)(ws + 786432);
    __bf16* h0  = (__bf16*)(ws + 17563648);
    __bf16* h1  = (__bf16*)(ws + 51118080);
    __bf16* h2  = h0;
    float*  logits = (float*)(ws + 51118080);

    const dim3 blk(256);

    cvt_f32_bf16<<<dim3((NROWS * DD + 255) / 256), blk, 0, stream>>>(batch, xb, NROWS * DD);
    transpose_w<<<dim3(2 * 128), blk, 0, stream>>>(w0, wT0, 128, 2 * 128 * 256);
    transpose_w<<<dim3(2 * 256), blk, 0, stream>>>(w1, wT1, 256, 2 * 256 * 256);
    transpose_w<<<dim3(2 * 256), blk, 0, stream>>>(w2, wT2, 256, 2 * 256 * 256);
    transpose_w<<<dim3(1 * 256), blk, 0, stream>>>(wo, wTo, 256, 1 * 256 * 256);

    const dim3 gconv(NROWS / 16);                        // 4096 workgroups
    conv_wmma<__bf16, 2, 128, true ><<<gconv, blk, 0, stream>>>(xb, wT0, b0, h0, 1);
    conv_wmma<__bf16, 2, 256, true ><<<gconv, blk, 0, stream>>>(h0, wT1, b1, h1, 2);
    conv_wmma<__bf16, 2, 256, true ><<<gconv, blk, 0, stream>>>(h1, wT2, b2, h2, 4);
    conv_wmma<float,  1, 256, false><<<gconv, blk, 0, stream>>>(h2, wTo, bo, logits, 1);

    softmax256<<<dim3(NROWS), blk, 0, stream>>>(logits);
    scan_fw<<<dim3(BB), blk, 0, stream>>>(logits, batch, probs, Wfin);
}
